// EdgeBiasedAttention_51376398795497
// MI455X (gfx1250) — matrix operand order repaired and checked by reading, not compiled, not run on hardware
//
#include <hip/hip_runtime.h>
#include <hip/hip_bf16.h>
#include <math.h>

// ---------------------------------------------------------------------------
// EdgeBiasedAttention for MI455X (gfx1250, wave32, WMMA bf16 w/ f32 accum)
// ---------------------------------------------------------------------------

typedef __attribute__((ext_vector_type(16))) __bf16 v16bf;
typedef __attribute__((ext_vector_type(8)))  __bf16 bf16x8;
typedef __attribute__((ext_vector_type(8)))  float  v8f;

#define N_NODES     8192
#define D_MODEL     512
#define N_HEADS     8
#define D_K         64
#define NUM_GRAPHS  64
#define NG          128
#define N_EDGES     262144

__device__ __forceinline__ v8f zero8() {
    v8f z;
#pragma unroll
    for (int i = 0; i < 8; ++i) z[i] = 0.0f;
    return z;
}

// A-fragment (16x32 bf16) from row-major source, rows row0..row0+15, K=kb..kb+31.
// Per ISA: lane<16 holds K in {kb..kb+7, kb+16..kb+23}; lane>=16 shifted by 8.
__device__ __forceinline__ v16bf load_a_frag(const __bf16* base, int ld, int row0, int kb) {
    int lane = threadIdx.x & 31;
    int half = lane >> 4;
    int r    = lane & 15;
    const __bf16* p = base + (size_t)(row0 + r) * ld + kb + half * 8;
    bf16x8 lo = *reinterpret_cast<const bf16x8*>(p);
    bf16x8 hi = *reinterpret_cast<const bf16x8*>(p + 16);
    v16bf v;
#pragma unroll
    for (int i = 0; i < 8; ++i) { v[i] = lo[i]; v[i + 8] = hi[i]; }
    return v;
}

// B-fragment (32x16 bf16) loaded from B^T (row-major N x K): column n = lane%16,
// lane<16 holds K in {kb..kb+7, kb+16..kb+23}; lane>=16 shifted by 8 (matches A).
__device__ __forceinline__ v16bf load_b_frag(const __bf16* bt, int ld, int col0, int kb) {
    int lane = threadIdx.x & 31;
    int half = lane >> 4;
    int c    = lane & 15;
    const __bf16* p = bt + (size_t)(col0 + c) * ld + kb + half * 8;
    bf16x8 lo = *reinterpret_cast<const bf16x8*>(p);
    bf16x8 hi = *reinterpret_cast<const bf16x8*>(p + 16);
    v16bf v;
#pragma unroll
    for (int i = 0; i < 8; ++i) { v[i] = lo[i]; v[i + 8] = hi[i]; }
    return v;
}

// ---------------- elementwise cast f32 -> bf16 ----------------
__global__ void __launch_bounds__(256)
cast_f32_bf16(const float* __restrict__ in, __bf16* __restrict__ out, int n) {
    int i = blockIdx.x * 256 + threadIdx.x;
    if (i < n) out[i] = (__bf16)in[i];
}

// ---------------- transpose 512x512 f32 -> bf16 (Wt[n][k] = W[k][n]) --------
__global__ void __launch_bounds__(256)
transpose512(const float* __restrict__ W, __bf16* __restrict__ Wt) {
    int idx = blockIdx.x * 256 + threadIdx.x;   // 262144 elements
    int n = idx >> 9, k = idx & 511;
    Wt[(size_t)n * 512 + k] = (__bf16)W[(size_t)k * 512 + n];
}

// ---------------- WMMA GEMM: Out = A(8192x512) @ Bt^T(512x512) + bias -------
// Fixed shape M=8192, N=512, K=512; K-loop fully unrolled with an explicit
// double buffer so next-step loads overlap current-step WMMAs.
// mode 0: fp32 row-major out;  mode 1: bf16 row-major out;
// mode 2: bf16 per-head-transposed V layout: Vt[(g*8+h)][d][local_node]
__global__ void __launch_bounds__(256)
wmma_gemm(const __bf16* __restrict__ A, const __bf16* __restrict__ Bt,
          const float* __restrict__ bias, void* __restrict__ Out, int mode) {
    constexpr int KDIM = 512;
    constexpr int NDIM = 512;
    int wave = threadIdx.x >> 5;
    int lane = threadIdx.x & 31;
    int half = lane >> 4, l16 = lane & 15;
    int m0 = blockIdx.x * 128 + wave * 16;
    int n0 = blockIdx.y * 64;

    v8f acc[4];
#pragma unroll
    for (int t = 0; t < 4; ++t) acc[t] = zero8();

    // prologue: load k-step 0
    v16bf a_cur = load_a_frag(A, KDIM, m0, 0);
    v16bf b_cur[4];
#pragma unroll
    for (int t = 0; t < 4; ++t) b_cur[t] = load_b_frag(Bt, KDIM, n0 + t * 16, 0);

#pragma unroll
    for (int step = 0; step < KDIM / 32; ++step) {
        v16bf a_nxt;
        v16bf b_nxt[4];
        if (step < KDIM / 32 - 1) {               // compile-time after unroll
            int kn = (step + 1) * 32;
            a_nxt = load_a_frag(A, KDIM, m0, kn); // issue loads BEFORE wmmas
#pragma unroll
            for (int t = 0; t < 4; ++t) b_nxt[t] = load_b_frag(Bt, KDIM, n0 + t * 16, kn);
        }
#pragma unroll
        for (int t = 0; t < 4; ++t)
            acc[t] = __builtin_amdgcn_wmma_f32_16x16x32_bf16(
                false, a_cur, false, b_cur[t], (short)0, acc[t], false, false);
        a_cur = a_nxt;
#pragma unroll
        for (int t = 0; t < 4; ++t) b_cur[t] = b_nxt[t];
    }

#pragma unroll
    for (int t = 0; t < 4; ++t) {
#pragma unroll
        for (int i = 0; i < 8; ++i) {
            int m = m0 + half * 8 + i;
            int n = n0 + t * 16 + l16;
            float v = acc[t][i] + bias[n];
            if (mode == 0) {
                ((float*)Out)[(size_t)m * NDIM + n] = v;
            } else if (mode == 1) {
                ((__bf16*)Out)[(size_t)m * NDIM + n] = (__bf16)v;
            } else {  // per-head transposed V: [(g*8+h)][d][ln]
                int g = m >> 7, ln = m & 127, h = n >> 6, d = n & 63;
                ((__bf16*)Out)[(((size_t)(g * 8 + h) * 64 + d) << 7) + ln] = (__bf16)v;
            }
        }
    }
}

// ---------------- edge MLP + scatter-add into dense bias [g][h][i][j] -------
__global__ void __launch_bounds__(256)
edge_mlp_scatter(const int* __restrict__ ei, const float* __restrict__ eattr,
                 const float* __restrict__ W1, const float* __restrict__ b1,
                 const float* __restrict__ W2, const float* __restrict__ b2,
                 float* __restrict__ Bias) {
    __shared__ float sW1[4 * 512];
    __shared__ float sW2[512 * 8];
    __shared__ float sb1[512];
    for (int i = threadIdx.x; i < 4 * 512; i += 256) sW1[i] = W1[i];
    for (int i = threadIdx.x; i < 512 * 8; i += 256) sW2[i] = W2[i];
    for (int i = threadIdx.x; i < 512;     i += 256) sb1[i] = b1[i];
    __syncthreads();

    int e = blockIdx.x * 256 + threadIdx.x;
    if (e >= N_EDGES) return;
    int s  = ei[e];
    int d  = ei[N_EDGES + e];
    int g  = s >> 7, ls = s & 127, ldn = d & 127;
    float a0 = eattr[e * 4 + 0], a1 = eattr[e * 4 + 1];
    float a2 = eattr[e * 4 + 2], a3 = eattr[e * 4 + 3];

    float acc[8];
#pragma unroll
    for (int h = 0; h < 8; ++h) acc[h] = 0.0f;

    for (int c = 0; c < 512; ++c) {
        float hc = fmaf(a0, sW1[c],
                   fmaf(a1, sW1[512 + c],
                   fmaf(a2, sW1[1024 + c],
                   fmaf(a3, sW1[1536 + c], sb1[c]))));
        hc = fmaxf(hc, 0.0f);
#pragma unroll
        for (int h = 0; h < 8; ++h) acc[h] = fmaf(hc, sW2[c * 8 + h], acc[h]);
    }
#pragma unroll
    for (int h = 0; h < 8; ++h)
        atomicAdd(&Bias[(((size_t)(g * 8 + h) * 128) + ls) * 128 + ldn], acc[h] + b2[h]);
}

// ---------------- fused per-(graph,head) attention --------------------------
// 8 waves/block; wave w owns score rows w*16..w*16+15 of the 128x128 block.
__global__ void __launch_bounds__(256)
attn_kernel(const __bf16* __restrict__ Qb, const __bf16* __restrict__ Kb,
            const __bf16* __restrict__ Vt, const float* __restrict__ Bias,
            __bf16* __restrict__ AO) {
    __shared__ __bf16 Plds[8 * 16 * 128];   // 32 KB: per-wave 16x128 P strip

    int gh = blockIdx.x;
    int g = gh >> 3, h = gh & 7;
    int wave = threadIdx.x >> 5;
    int lane = threadIdx.x & 31;
    int half = lane >> 4, l16 = lane & 15;
    int m0 = wave * 16;

    const __bf16* Q  = Qb + (size_t)g * 128 * 512 + h * 64;   // ld 512
    const __bf16* K  = Kb + (size_t)g * 128 * 512 + h * 64;   // ld 512
    const __bf16* V  = Vt + (size_t)gh * 64 * 128;            // d x node, ld 128
    const float*  Bh = Bias + (size_t)gh * 128 * 128;         // [i][j]

    // S = Q @ K^T  -- batch all B-frag loads per k-step, then the WMMA chain
    v8f acc[8];
#pragma unroll
    for (int nt = 0; nt < 8; ++nt) acc[nt] = zero8();
#pragma unroll
    for (int kt = 0; kt < 64; kt += 32) {
        v16bf a = load_a_frag(Q, 512, m0, kt);
        v16bf b[8];
#pragma unroll
        for (int nt = 0; nt < 8; ++nt) b[nt] = load_b_frag(K, 512, nt * 16, kt);
#pragma unroll
        for (int nt = 0; nt < 8; ++nt)
            acc[nt] = __builtin_amdgcn_wmma_f32_16x16x32_bf16(
                false, a, false, b[nt], (short)0, acc[nt], false, false);
    }

    // scale + edge bias
#pragma unroll
    for (int nt = 0; nt < 8; ++nt) {
#pragma unroll
        for (int i = 0; i < 8; ++i) {
            int m = m0 + half * 8 + i;
            int n = nt * 16 + l16;
            acc[nt][i] = acc[nt][i] * 0.125f + Bh[m * 128 + n];
        }
    }

    // softmax over j: each row lives in 16 lanes of one half-wave (+8 regs each)
#pragma unroll
    for (int i = 0; i < 8; ++i) {
        float mx = -3.0e38f;
#pragma unroll
        for (int nt = 0; nt < 8; ++nt) mx = fmaxf(mx, acc[nt][i]);
#pragma unroll
        for (int dlt = 1; dlt < 16; dlt <<= 1) mx = fmaxf(mx, __shfl_xor(mx, dlt, 32));
        float sm = 0.0f;
#pragma unroll
        for (int nt = 0; nt < 8; ++nt) {
            float e = __expf(acc[nt][i] - mx);
            acc[nt][i] = e;
            sm += e;
        }
#pragma unroll
        for (int dlt = 1; dlt < 16; dlt <<= 1) sm += __shfl_xor(sm, dlt, 32);
        float inv = 1.0f / sm;
#pragma unroll
        for (int nt = 0; nt < 8; ++nt) acc[nt][i] *= inv;
    }

    // C-layout -> A-layout transpose through LDS (bf16)
    __bf16* Pw = Plds + wave * 16 * 128;
#pragma unroll
    for (int nt = 0; nt < 8; ++nt) {
#pragma unroll
        for (int i = 0; i < 8; ++i)
            Pw[(half * 8 + i) * 128 + nt * 16 + l16] = (__bf16)acc[nt][i];
    }
    __syncthreads();

    // O = P @ V  (B^T = Vt rows are contiguous); batch B-frag loads per k-step
    v8f oacc[4];
#pragma unroll
    for (int nt = 0; nt < 4; ++nt) oacc[nt] = zero8();
#pragma unroll
    for (int jt = 0; jt < 128; jt += 32) {
        v16bf a = load_a_frag(Pw, 128, 0, jt);
        v16bf b[4];
#pragma unroll
        for (int nt = 0; nt < 4; ++nt) b[nt] = load_b_frag(V, 128, nt * 16, jt);
#pragma unroll
        for (int nt = 0; nt < 4; ++nt)
            oacc[nt] = __builtin_amdgcn_wmma_f32_16x16x32_bf16(
                false, a, false, b[nt], (short)0, oacc[nt], false, false);
    }

    // store into AO row-major [node][h*64+d] (bf16) for the final Wo GEMM
#pragma unroll
    for (int nt = 0; nt < 4; ++nt) {
#pragma unroll
        for (int i = 0; i < 8; ++i) {
            int m = m0 + half * 8 + i;
            int dcol = nt * 16 + l16;
            AO[(size_t)(g * 128 + m) * 512 + h * 64 + dcol] = (__bf16)oacc[nt][i];
        }
    }
}

// ---------------------------------------------------------------------------
extern "C" void kernel_launch(void* const* d_in, const int* in_sizes, int n_in,
                              void* d_out, int out_size, void* d_ws, size_t ws_size,
                              hipStream_t stream) {
    (void)in_sizes; (void)n_in; (void)out_size; (void)ws_size;
    const float* x     = (const float*)d_in[0];
    const int*   ei    = (const int*)d_in[1];
    const float* eattr = (const float*)d_in[2];
    // d_in[3] = batch, d_in[4] = n_g : unused (constants hardcoded)
    const float* Wq = (const float*)d_in[5];  const float* bq = (const float*)d_in[6];
    const float* Wk = (const float*)d_in[7];  const float* bk = (const float*)d_in[8];
    const float* Wv = (const float*)d_in[9];  const float* bv = (const float*)d_in[10];
    const float* Wo = (const float*)d_in[11]; const float* bo = (const float*)d_in[12];
    const float* W1 = (const float*)d_in[13]; const float* b1 = (const float*)d_in[14];
    const float* W2 = (const float*)d_in[15]; const float* b2 = (const float*)d_in[16];

    char* ws = (char*)d_ws;
    size_t off = 0;
    const size_t MAT_BF = (size_t)N_NODES * D_MODEL * sizeof(__bf16);   // 8 MB
    const size_t W_BF   = (size_t)D_MODEL * D_MODEL * sizeof(__bf16);   // 512 KB
    __bf16* x_bf  = (__bf16*)(ws + off); off += MAT_BF;
    __bf16* Wqt   = (__bf16*)(ws + off); off += W_BF;
    __bf16* Wkt   = (__bf16*)(ws + off); off += W_BF;
    __bf16* Wvt   = (__bf16*)(ws + off); off += W_BF;
    __bf16* Wot   = (__bf16*)(ws + off); off += W_BF;
    __bf16* Q_bf  = (__bf16*)(ws + off); off += MAT_BF;
    __bf16* K_bf  = (__bf16*)(ws + off); off += MAT_BF;
    __bf16* Vt_bf = (__bf16*)(ws + off); off += MAT_BF;
    __bf16* AO_bf = (__bf16*)(ws + off); off += MAT_BF;
    float*  Bias  = (float*)(ws + off);
    const size_t BIAS_BYTES = (size_t)NUM_GRAPHS * N_HEADS * NG * NG * sizeof(float); // 33.5 MB

    hipMemsetAsync(Bias, 0, BIAS_BYTES, stream);

    // casts / weight transposes
    {
        int n = N_NODES * D_MODEL;
        cast_f32_bf16<<<(n + 255) / 256, 256, 0, stream>>>(x, x_bf, n);
        transpose512<<<1024, 256, 0, stream>>>(Wq, Wqt);
        transpose512<<<1024, 256, 0, stream>>>(Wk, Wkt);
        transpose512<<<1024, 256, 0, stream>>>(Wv, Wvt);
        transpose512<<<1024, 256, 0, stream>>>(Wo, Wot);
    }

    dim3 ggrid(N_NODES / 128, D_MODEL / 64);
    wmma_gemm<<<ggrid, 256, 0, stream>>>(x_bf, Wqt, bq, Q_bf,  1);
    wmma_gemm<<<ggrid, 256, 0, stream>>>(x_bf, Wkt, bk, K_bf,  1);
    wmma_gemm<<<ggrid, 256, 0, stream>>>(x_bf, Wvt, bv, Vt_bf, 2);

    edge_mlp_scatter<<<N_EDGES / 256, 256, 0, stream>>>(ei, eattr, W1, b1, W2, b2, Bias);

    attn_kernel<<<NUM_GRAPHS * N_HEADS, 256, 0, stream>>>(Q_bf, K_bf, Vt_bf, Bias, AO_bf);

    wmma_gemm<<<ggrid, 256, 0, stream>>>(AO_bf, Wot, bo, d_out, N_NODES / 128 ? 0 : 0);
}
